// MoERegressor_25202868093195
// MI455X (gfx1250) — matrix operand order repaired and checked by reading, not compiled, not run on hardware
//
#include <hip/hip_runtime.h>
#include <math.h>

#define DIM 256
#define NE 16
#define EH 128
#define RH 64
#define NTOK 32768
#define MBLK 64
#define THREADS 256

// padded LDS row lengths (in halves): +8 halves = +16B keeps 16B alignment,
// gives 528B (=132 dword) row stride -> bank stride 4, conflict-free b128 frags
#define XP  (DIM + 8)
#define W1P (DIM + 8)
#define W2P (EH + 8)
#define HP  (EH + 8)

#define WS_BYTES_NEEDED (2u * 1024u * 1024u)   // 2x 1MB transposed f16 weights

typedef __attribute__((ext_vector_type(16))) _Float16 v16h;
typedef __attribute__((ext_vector_type(8)))  _Float16 v8h;
typedef __attribute__((ext_vector_type(8)))  float    v8f;
typedef __attribute__((ext_vector_type(4)))  unsigned int v4u;
typedef __attribute__((ext_vector_type(8)))  int v8i;
typedef __attribute__((ext_vector_type(4)))  int v4i;

#if defined(__has_builtin)
#if __has_builtin(__builtin_amdgcn_tensor_load_to_lds)
#define TDM_OK 1
#endif
#endif
#ifndef TDM_OK
#define TDM_OK 0
#endif

// A-fragment (16x32 f16): lane L (r=L&15,hi=L>>4) holds row M=r,
// halves 0..7 = K kbase..kbase+7, halves 8..15 = K kbase+16..kbase+23, kbase=kt*32+hi*8
__device__ __forceinline__ v16h ld_frag_a(const _Float16* rowp, int kbase) {
  union { v16h v; v8h h[2]; } u;
  u.h[0] = *(const v8h*)(rowp + kbase);
  u.h[1] = *(const v8h*)(rowp + kbase + 16);
  return u.v;
}
// B-fragment (32x16 f16): lane L holds col N=L&15, halves 0..7 = K kbase..kbase+7, halves 8..15 = K kbase+16..kbase+23, kbase=kt*32+hi*8
__device__ __forceinline__ v16h ld_frag_b(const _Float16* rowp, int kbase) {
  union { v16h v; v8h h[2]; } u;
  u.h[0] = *(const v8h*)(rowp + kbase);
  u.h[1] = *(const v8h*)(rowp + kbase + 16);
  return u.v;
}

#if TDM_OK
// 2-D TDM load: rows x (tdim0_dw dwords) from global -> LDS, inserting
// pad_amount=4 dwords (16B) after every 2^(pad_code+1) dwords (one row).
// D# group0: count=1, lds_addr, 57b global addr, type=2 (image).
// D# group1: data_size=4B, pad_enable, pad_interval=pad_code, pad_amount=3 (->4dw),
//            tensor_dim0=tile_dim0=stride0=tdim0_dw, tensor_dim1=tile_dim1=rows.
// 6-arg builtin (clang-23 / therock lane): groups 2/3 + VADDR4 slot zeroed (2D tensor).
__device__ __forceinline__ void tdm_load_2d(unsigned int lds_off, const void* gptr,
                                            int tdim0_dw, int rows, int pad_code) {
  unsigned long long ga = (unsigned long long)(uintptr_t)gptr;
  v4u g0;
  g0.x = 1u;                                   // count=1 (valid user descriptor)
  g0.y = lds_off;                              // lds_addr
  g0.z = (unsigned int)ga;                     // global_addr[31:0]
  g0.w = (unsigned int)(ga >> 32) | (2u << 30);// global_addr[56:32] | type=2
  v8i g1;
  g1[0] = (int)((2u << 16) | (1u << 20) | ((unsigned)pad_code << 22) | (3u << 25));
  g1[1] = (int)((unsigned)(tdim0_dw & 0xFFFF) << 16);            // tensor_dim0 lo
  g1[2] = (int)(((unsigned)tdim0_dw >> 16) | ((unsigned)(rows & 0xFFFF) << 16));
  g1[3] = (int)(((unsigned)rows >> 16) | ((unsigned)(tdim0_dw & 0xFFFF) << 16)); // tile_dim0
  g1[4] = (int)(unsigned)rows;                 // tile_dim1 (tile_dim2=0 unused)
  g1[5] = (int)(unsigned)tdim0_dw;             // tensor_dim0_stride[31:0]
  g1[6] = 0;
  g1[7] = 0;
  v4i gz4 = {0, 0, 0, 0};
  v8i gz8 = {0, 0, 0, 0, 0, 0, 0, 0};
  __builtin_amdgcn_tensor_load_to_lds(g0, g1, gz4, gz4, gz8, 0);
}
#endif

// convert + transpose expert weights to f16 in workspace:
// w1h[e][n][k] (k contiguous, 512B rows), w2h[e][d][h] (h contiguous, 256B rows)
__global__ __launch_bounds__(THREADS)
void prep_weights(const float* __restrict__ e_w1, const float* __restrict__ e_w2,
                  _Float16* __restrict__ w1h, _Float16* __restrict__ w2h) {
  int idx = blockIdx.x * THREADS + threadIdx.x;   // 0 .. 16*256*128-1
  int e = idx >> 15, r = idx & 32767;
  int k = r >> 7, n = r & (EH - 1);               // e_w1[e][k][n]
  w1h[(size_t)e * (DIM * EH) + n * DIM + k] = (_Float16)e_w1[idx];
  int h = r >> 8, d = r & (DIM - 1);              // e_w2[e][h][d]
  w2h[(size_t)e * (DIM * EH) + d * EH + h] = (_Float16)e_w2[idx];
}

template <bool USE_TDM>
__global__ __launch_bounds__(THREADS, 1)
void moe_fused_kernel(const float* __restrict__ x,
                      const float* __restrict__ r_w1, const float* __restrict__ r_b1,
                      const float* __restrict__ r_w2, const float* __restrict__ r_b2,
                      const float* __restrict__ e_w1, const float* __restrict__ e_b1,
                      const float* __restrict__ e_w2, const float* __restrict__ e_b2,
                      const _Float16* __restrict__ w1h, const _Float16* __restrict__ w2h,
                      float* __restrict__ out)
{
  __shared__ _Float16 x_lds[MBLK][XP];   // token tile, f16
  __shared__ _Float16 w1t[EH][W1P];      // w1 transposed: w1t[n][k]
  __shared__ _Float16 w2t[DIM][W2P];     // w2 transposed: w2t[d][h]
  __shared__ _Float16 h_lds[MBLK][HP];   // tanh hidden (aliased as router hr f32)
  __shared__ float g_lds[MBLK][NE];      // top-2 renormalized gates
  __shared__ float logit_lds[MBLK][NE];
  __shared__ float b1_all[NE][EH];       // all expert biases, loaded once
  __shared__ float b2_all[NE][DIM];

  const int tid  = threadIdx.x;
  const int lane = tid & 31;
  const int wave = tid >> 5;
  const int blk  = blockIdx.x;
  const float* xg = x + (size_t)blk * MBLK * DIM;

#if TDM_OK
  unsigned int w1t_off = 0, w2t_off = 0;
  if constexpr (USE_TDM) {
    w1t_off = (unsigned int)(uintptr_t)&w1t[0][0];  // generic LDS ptr low 32b = LDS offset
    w2t_off = (unsigned int)(uintptr_t)&w2t[0][0];
    if (wave == 0)  // prefetch w1t[0]: overlaps x staging + router
      tdm_load_2d(w1t_off, w1h, DIM / 2, EH, 6);
  }
#endif

  // ---- stage x tile as f16; preload all biases ----
  for (int i = tid; i < MBLK * DIM; i += THREADS) {
    int n = i >> 8, k = i & (DIM - 1);
    x_lds[n][k] = (_Float16)xg[i];
  }
  for (int i = tid; i < NE * EH; i += THREADS) ((float*)b1_all)[i] = e_b1[i];
  for (int i = tid; i < NE * DIM; i += THREADS) ((float*)b2_all)[i] = e_b2[i];

  // ---- router (fp32, exact): hr = relu(x@r_w1+b1) stored in h_lds region ----
  float (*hr)[RH] = reinterpret_cast<float(*)[RH]>(&h_lds[0][0]);
  {
    const int n = tid >> 2, q = tid & 3;     // 64 tokens x 4 groups of 16 hidden
    float acc[16];
    #pragma unroll
    for (int j = 0; j < 16; ++j) acc[j] = 0.f;
    const float* xrow = xg + n * DIM;
    for (int k = 0; k < DIM; ++k) {
      float xv = xrow[k];
      const float* wr = r_w1 + k * RH + q * 16;
      #pragma unroll
      for (int j = 0; j < 16; ++j) acc[j] += xv * wr[j];
    }
    #pragma unroll
    for (int j = 0; j < 16; ++j)
      hr[n][q * 16 + j] = fmaxf(acc[j] + r_b1[q * 16 + j], 0.f);
  }
  __syncthreads();
  {
    const int n = tid >> 2, q = tid & 3;     // 4 logits per thread
    float a4[4] = {0.f, 0.f, 0.f, 0.f};
    for (int k = 0; k < RH; ++k) {
      float hv = hr[n][k];
      #pragma unroll
      for (int j = 0; j < 4; ++j) a4[j] += hv * r_w2[k * NE + q * 4 + j];
    }
    #pragma unroll
    for (int j = 0; j < 4; ++j)
      logit_lds[n][q * 4 + j] = a4[j] + r_b2[q * 4 + j];
  }
  __syncthreads();
  if (tid < MBLK) {                          // softmax + top-2 per token
    const int n = tid;
    float l[NE], wv[NE];
    float m = -3.0e38f;
    #pragma unroll
    for (int e = 0; e < NE; ++e) { l[e] = logit_lds[n][e]; m = fmaxf(m, l[e]); }
    float s = 0.f;
    #pragma unroll
    for (int e = 0; e < NE; ++e) { wv[e] = expf(l[e] - m); s += wv[e]; }
    float inv = 1.f / s;
    float* outw = out + (size_t)NTOK * DIM + (size_t)(blk * MBLK + n) * NE;
    #pragma unroll
    for (int e = 0; e < NE; ++e) { wv[e] *= inv; outw[e] = wv[e]; }
    int i1 = 0;
    #pragma unroll
    for (int e = 1; e < NE; ++e) if (wv[e] > wv[i1]) i1 = e;       // stable argmax
    int i2 = (i1 == 0) ? 1 : 0;
    #pragma unroll
    for (int e = 0; e < NE; ++e) if (e != i1 && wv[e] > wv[i2]) i2 = e;
    float denom = wv[i1] + wv[i2] + 1e-8f;
    #pragma unroll
    for (int e = 0; e < NE; ++e)
      g_lds[n][e] = (e == i1 || e == i2) ? wv[e] / denom : 0.f;
  }

  // ---- expert loop: TDM-pipelined weight staging + WMMA GEMM1/GEMM2 ----
  const int r  = lane & 15;
  const int hi = lane >> 4;
  const int m0  = (wave >> 1) * 16;    // 4 M-tiles, 2 waves each
  const int nb1 = (wave & 1) * 4;      // GEMM1: 4 N-tiles of 128
  const int nb2 = (wave & 1) * 8;      // GEMM2: 8 N-tiles of 256

  v8f yacc[8] = {};

  #pragma unroll 1
  for (int e = 0; e < NE; ++e) {
    bool staged = false;
#if TDM_OK
    if constexpr (USE_TDM) {
      staged = true;
      if (wave == 0) __builtin_amdgcn_s_wait_tensorcnt(0);  // w1t[e] landed
      __syncthreads();                                      // barrier A
      if (wave == 0)                                        // DMA w2t[e] during GEMM1[e]
        tdm_load_2d(w2t_off, w2h + (size_t)e * (DIM * EH), EH / 2, DIM, 5);
    }
#endif
    if (!staged) {  // synchronous fallback (no TDM / no workspace)
      __syncthreads();
      const float* w1g = e_w1 + (size_t)e * DIM * EH;
      for (int i = tid; i < DIM * EH; i += THREADS) {
        int k = i >> 7, n = i & (EH - 1);
        w1t[n][k] = (_Float16)w1g[i];
      }
      const float* w2g = e_w2 + (size_t)e * EH * DIM;
      for (int i = tid; i < EH * DIM; i += THREADS) {
        int h = i >> 8, d = i & (DIM - 1);
        w2t[d][h] = (_Float16)w2g[i];
      }
      __syncthreads();
    }

    // GEMM1: h = tanh(x @ w1 + b1), M=64 K=256 N=128
    v8f c1[4] = {};
    #pragma unroll
    for (int kt = 0; kt < DIM / 32; ++kt) {
      v16h af = ld_frag_a(&x_lds[m0 + r][0], kt * 32 + hi * 8);
      #pragma unroll
      for (int nt = 0; nt < 4; ++nt) {
        v16h bf = ld_frag_b(&w1t[(nb1 + nt) * 16 + r][0], kt * 32 + hi * 8);
        c1[nt] = __builtin_amdgcn_wmma_f32_16x16x32_f16(
                     false, af, false, bf, (short)0, c1[nt], false, false);
      }
    }
    #pragma unroll
    for (int nt = 0; nt < 4; ++nt) {
      int cn = (nb1 + nt) * 16 + r;            // D-layout: col = lane&15
      float b = b1_all[e][cn];
      #pragma unroll
      for (int i = 0; i < 8; ++i)              // row = m0 + hi*8 + i
        h_lds[m0 + hi * 8 + i][cn] = (_Float16)tanhf(c1[nt][i] + b);
    }
#if TDM_OK
    if constexpr (USE_TDM) {
      if (wave == 0) __builtin_amdgcn_s_wait_tensorcnt(0);  // w2t[e] landed
    }
#endif
    __syncthreads();                                        // barrier B
#if TDM_OK
    if constexpr (USE_TDM) {
      if (wave == 0 && e + 1 < NE)                          // DMA w1t[e+1] during GEMM2[e]
        tdm_load_2d(w1t_off, w1h + (size_t)(e + 1) * (DIM * EH), DIM / 2, EH, 6);
    }
#endif

    float gv[8];
    #pragma unroll
    for (int i = 0; i < 8; ++i) gv[i] = g_lds[m0 + hi * 8 + i][e];

    // GEMM2: y_e = h @ w2 + b2, M=64 K=128 N=256; accumulate g*y_e
    v8f c2[8] = {};
    #pragma unroll
    for (int kt = 0; kt < EH / 32; ++kt) {
      v16h af = ld_frag_a(&h_lds[m0 + r][0], kt * 32 + hi * 8);
      #pragma unroll
      for (int nt = 0; nt < 8; ++nt) {
        v16h bf = ld_frag_b(&w2t[(nb2 + nt) * 16 + r][0], kt * 32 + hi * 8);
        c2[nt] = __builtin_amdgcn_wmma_f32_16x16x32_f16(
                     false, af, false, bf, (short)0, c2[nt], false, false);
      }
    }
    #pragma unroll
    for (int nt = 0; nt < 8; ++nt) {
      float b = b2_all[e][(nb2 + nt) * 16 + r];
      #pragma unroll
      for (int i = 0; i < 8; ++i)
        yacc[nt][i] += gv[i] * (c2[nt][i] + b);
    }
  }

  // ---- store y_pred ----
  #pragma unroll
  for (int nt = 0; nt < 8; ++nt) {
    int col = (nb2 + nt) * 16 + r;
    #pragma unroll
    for (int i = 0; i < 8; ++i) {
      size_t row = (size_t)blk * MBLK + m0 + hi * 8 + i;
      out[row * DIM + col] = yacc[nt][i];
    }
  }
}

extern "C" void kernel_launch(void* const* d_in, const int* in_sizes, int n_in,
                              void* d_out, int out_size, void* d_ws, size_t ws_size,
                              hipStream_t stream) {
  (void)in_sizes; (void)n_in; (void)out_size;
  const float* x    = (const float*)d_in[0];
  const float* r_w1 = (const float*)d_in[1];
  const float* r_b1 = (const float*)d_in[2];
  const float* r_w2 = (const float*)d_in[3];
  const float* r_b2 = (const float*)d_in[4];
  const float* e_w1 = (const float*)d_in[5];
  const float* e_b1 = (const float*)d_in[6];
  const float* e_w2 = (const float*)d_in[7];
  const float* e_b2 = (const float*)d_in[8];
  float* out = (float*)d_out;

  _Float16* w1h = (_Float16*)d_ws;
  _Float16* w2h = w1h + (size_t)NE * DIM * EH;   // +1MB

  const bool use_tdm = TDM_OK && d_ws != nullptr && ws_size >= WS_BYTES_NEEDED;
  dim3 grid(NTOK / MBLK);   // 512 workgroups

  if (use_tdm) {
    prep_weights<<<dim3((NE * DIM * EH) / THREADS), THREADS, 0, stream>>>(e_w1, e_w2, w1h, w2h);
    moe_fused_kernel<true><<<grid, THREADS, 0, stream>>>(
        x, r_w1, r_b1, r_w2, r_b2, e_w1, e_b1, e_w2, e_b2, w1h, w2h, out);
  } else {
    moe_fused_kernel<false><<<grid, THREADS, 0, stream>>>(
        x, r_w1, r_b1, r_w2, r_b2, e_w1, e_b1, e_w2, e_b2, w1h, w2h, out);
  }
}
